// TriangleProteinToCompound_v2_56590489092295
// MI455X (gfx1250) — matrix operand order repaired and checked by reading, not compiled, not run on hardware
//
#include <hip/hip_runtime.h>
#include <hip/hip_bf16.h>
#include <math.h>
#include <stdint.h>

// ---------------------------------------------------------------------------
// TriangleMultiplication (protein→compound) for MI455X / gfx1250.
// bf16 WMMA (v_wmma_f32_16x16x32_bf16) with f32 accumulation everywhere.
// Triangle einsums are LDS-tiled per WGP with double-buffered async
// global->LDS staging (GLOBAL_LOAD_ASYNC_TO_LDS_B128 / ASYNCcnt).
// ---------------------------------------------------------------------------

typedef __bf16 bf16_t;
typedef __attribute__((ext_vector_type(16))) __bf16 bf16x16;
typedef __attribute__((ext_vector_type(8)))  __bf16 bf16x8;
typedef __attribute__((ext_vector_type(8)))  float  f32x8;

#define LN_EPS 1e-5f
#define USE_ASYNC_LDS 1

__device__ __forceinline__ float wave_sum32(float s) {
#pragma unroll
  for (int off = 16; off > 0; off >>= 1) s += __shfl_xor(s, off, 32);
  return s;
}

// A operand: 16x32 bf16 tile, row-major source with leading dim `ld`.
// lane m (0..15) holds row m: K = 0..15;
// lane m+16 holds row m: K = 16..31.
__device__ __forceinline__ bf16x16 load_amat(const bf16_t* __restrict__ base, int ld,
                                             int row0, int k0, int lane) {
  int m = lane & 15, h = lane >> 4;
  const bf16_t* p = base + (size_t)(row0 + m) * ld + k0 + 16 * h;
  bf16x16 a;
  ((bf16x8*)&a)[0] = *(const bf16x8*)(p);       // K block 0 (8 elems, 16B)
  ((bf16x8*)&a)[1] = *(const bf16x8*)(p + 8);   // K block 1
  return a;
}

// B operand: 32x16 bf16 tile from an [N][K] (K-contiguous) source.
// lane n (0..15) holds column n, K=0..15; lane n+16 holds column n, K=16..31.
__device__ __forceinline__ bf16x16 load_bmat(const bf16_t* __restrict__ base, int ld,
                                             int col0, int k0, int lane) {
  int n = lane & 15, h = lane >> 4;
  const bf16_t* p = base + (size_t)(col0 + n) * ld + k0 + 16 * h;
  bf16x16 b;
  ((bf16x8*)&b)[0] = *(const bf16x8*)(p);
  ((bf16x8*)&b)[1] = *(const bf16x8*)(p + 8);
  return b;
}

__device__ __forceinline__ f32x8 wmma_bf16(bf16x16 a, bf16x16 b, f32x8 c) {
  return __builtin_amdgcn_wmma_f32_16x16x32_bf16(false, a, false, b, (short)0, c,
                                                 false, false);
}

__device__ __forceinline__ float sigmoidf_dev(float x) {
  return 1.f / (1.f + __expf(-x));
}

#if USE_ASYNC_LDS
// 16B per lane, global -> LDS, tracked by ASYNCcnt (no VGPR round trip).
__device__ __forceinline__ void async_copy_b128(uint32_t lds_off, const void* gaddr) {
  asm volatile("global_load_async_to_lds_b128 %0, %1, off"
               :
               : "v"(lds_off), "v"((unsigned long long)(uintptr_t)gaddr)
               : "memory");
}
__device__ __forceinline__ void wait_asynccnt0() {
  asm volatile("s_wait_asynccnt 0x0" ::: "memory");
}
#endif

// ---------------------------------------------------------------------------
// LayerNorm over last dim 256 -> bf16. One wave per row.
// ---------------------------------------------------------------------------
__global__ void ln_rows256_bf16(const float* __restrict__ x, const float* __restrict__ w,
                                bf16_t* __restrict__ y, int nrows) {
  int wid = (int)((blockIdx.x * blockDim.x + threadIdx.x) >> 5);
  int lane = threadIdx.x & 31;
  if (wid >= nrows) return;
  const float* row = x + (size_t)wid * 256;
  float v[8], s = 0.f;
#pragma unroll
  for (int i = 0; i < 8; ++i) { v[i] = row[i * 32 + lane]; s += v[i]; }
  float mean = wave_sum32(s) * (1.f / 256.f);
  float q = 0.f;
#pragma unroll
  for (int i = 0; i < 8; ++i) { float d = v[i] - mean; q += d * d; }
  float rstd = rsqrtf(wave_sum32(q) * (1.f / 256.f) + LN_EPS);
  bf16_t* o = y + (size_t)wid * 256;
#pragma unroll
  for (int i = 0; i < 8; ++i)
    o[i * 32 + lane] = (bf16_t)((v[i] - mean) * rstd * w[i * 32 + lane]);
}

// LayerNorm over last dim 128 -> bf16. One wave per row.
__global__ void ln_rows128_bf16(const float* __restrict__ x, const float* __restrict__ w,
                                bf16_t* __restrict__ y, int nrows) {
  int wid = (int)((blockIdx.x * blockDim.x + threadIdx.x) >> 5);
  int lane = threadIdx.x & 31;
  if (wid >= nrows) return;
  const float* row = x + (size_t)wid * 128;
  float v[4], s = 0.f;
#pragma unroll
  for (int i = 0; i < 4; ++i) { v[i] = row[i * 32 + lane]; s += v[i]; }
  float mean = wave_sum32(s) * (1.f / 128.f);
  float q = 0.f;
#pragma unroll
  for (int i = 0; i < 4; ++i) { float d = v[i] - mean; q += d * d; }
  float rstd = rsqrtf(wave_sum32(q) * (1.f / 128.f) + LN_EPS);
  bf16_t* o = y + (size_t)wid * 128;
#pragma unroll
  for (int i = 0; i < 4; ++i)
    o[i * 32 + lane] = (bf16_t)((v[i] - mean) * rstd * w[i * 32 + lane]);
}

// ---------------------------------------------------------------------------
// Convert f32 weight [K][N] -> bf16 transposed [N][K] (K-contiguous, B-operand
// friendly).
// ---------------------------------------------------------------------------
__global__ void cvt_w_transpose(const float* __restrict__ w, bf16_t* __restrict__ wt,
                                int K, int N) {
  int t = blockIdx.x * blockDim.x + threadIdx.x;
  if (t >= K * N) return;
  int n = t / K, k = t - n * K;
  wt[t] = (bf16_t)w[(size_t)k * N + n];
}

// ---------------------------------------------------------------------------
// Gated projection: y[r][c] = sigmoid(X[r]·G[:,c]) * (X[r]·W[:,c] + b[c]) * m[r]
// X: bf16 [R][256]; Gt/Wt: bf16 [128][256] ([N][K]).
// Output scattered channel-major: dst[c*R + remap(r)],
// remap(r) = (r % rmod)*rmul + (r / rmod)*rdiv  (identity: rmod=R,rmul=1,rdiv=0)
// One wave per 16-row tile; the A operand stays register-resident across the
// n-tile loop (compiler hoists it into 64 VGPRs).
// ---------------------------------------------------------------------------
__global__ void __launch_bounds__(256)
gated_proj(const bf16_t* __restrict__ X, const bf16_t* __restrict__ Gt,
           const bf16_t* __restrict__ Wt, const float* __restrict__ bias,
           const float* __restrict__ mask, bf16_t* __restrict__ dst,
           int R, int rmod, int rmul, int rdiv) {
  int wid = (int)((blockIdx.x * blockDim.x + threadIdx.x) >> 5);
  int lane = threadIdx.x & 31;
  int row0 = wid * 16;
  if (row0 >= R) return;
  int nlo = lane & 15, h = lane >> 4;
#pragma unroll 1
  for (int nt = 0; nt < 8; ++nt) {
    int n0 = nt * 16;
    f32x8 accg = (f32x8)0.f;
    f32x8 accw = (f32x8)0.f;
#pragma unroll
    for (int kc = 0; kc < 8; ++kc) {
      bf16x16 a  = load_amat(X, 256, row0, kc * 32, lane);
      bf16x16 bg = load_bmat(Gt, 256, n0, kc * 32, lane);
      bf16x16 bw = load_bmat(Wt, 256, n0, kc * 32, lane);
      accg = wmma_bf16(a, bg, accg);
      accw = wmma_bf16(a, bw, accw);
    }
    float bcol = bias[n0 + nlo];
#pragma unroll
    for (int v = 0; v < 8; ++v) {
      int rr = row0 + v + 8 * h;  // D layout: lane&15 = col, vgpr + 8*(lane>>4) = row
      float val = sigmoidf_dev(accg[v]) * (accw[v] + bcol);
      if (mask) val *= mask[rr];
      size_t rd = (size_t)(rr % rmod) * rmul + (size_t)(rr / rmod) * rdiv;
      dst[(size_t)(n0 + nlo) * R + rd] = (bf16_t)val;
    }
  }
}

// ---------------------------------------------------------------------------
// Dense projection, two epilogue modes:
//   mode 0: out = sigmoid(X@W + b)                (the `g` gate, f32 out)
//   mode 1: out = gmul * (X@W + b) * mask         (final output projection)
// X: bf16 [R][K]; Wt: bf16 [N][K]; out: f32 [R][N].
// ---------------------------------------------------------------------------
__global__ void __launch_bounds__(256)
dense_proj(const bf16_t* __restrict__ X, const bf16_t* __restrict__ Wt,
           const float* __restrict__ bias, const float* __restrict__ gmul,
           const float* __restrict__ mask, float* __restrict__ out,
           int R, int N, int K, int mode) {
  int wid = (int)((blockIdx.x * blockDim.x + threadIdx.x) >> 5);
  int lane = threadIdx.x & 31;
  int row0 = wid * 16;
  if (row0 >= R) return;
  int nlo = lane & 15, h = lane >> 4;
  int ntiles = N >> 4, kch = K >> 5;
#pragma unroll 1
  for (int nt = 0; nt < ntiles; ++nt) {
    int n0 = nt * 16;
    f32x8 acc = (f32x8)0.f;
#pragma unroll 4
    for (int kc = 0; kc < kch; ++kc) {
      bf16x16 a = load_amat(X, K, row0, kc * 32, lane);
      bf16x16 b = load_bmat(Wt, K, n0, kc * 32, lane);
      acc = wmma_bf16(a, b, acc);
    }
    float bcol = bias[n0 + nlo];
#pragma unroll
    for (int v = 0; v < 8; ++v) {
      int rr = row0 + v + 8 * h;
      float val = acc[v] + bcol;
      if (mode == 0) {
        val = sigmoidf_dev(val);
      } else {
        val *= gmul[(size_t)rr * N + n0 + nlo];
        val *= mask[rr];
      }
      out[(size_t)rr * N + n0 + nlo] = val;
    }
  }
}

// ---------------------------------------------------------------------------
// Fused triangle einsums, LDS-tiled + double-buffered async staging.
// Per channel c:
//   bsum[i][j][c] = sum_k ppA[c][i][k]*ab1B[c][j][k]   (K = 512, chunks 0..15)
//                 + sum_k ab2A[c][i][k]*cpB[c][j][k]   (K = 128, chunks 16..19)
// One workgroup = (channel, 64-row i-block) x all 128 j (8 waves x 4 tiles).
// Pipeline: at chunk kc, async-copy chunk kc+1 into the other LDS buffer
// (ASYNCcnt), run 4 WMMAs from the ready buffer, then s_wait_asynccnt 0 +
// barrier. The copy overlaps the full compute phase of the current chunk.
// Operand reuse vs. naive per-wave version: A x8, B x4.
// ---------------------------------------------------------------------------
__global__ void __launch_bounds__(256)
fused_block_gemm(const bf16_t* __restrict__ ppA,   // [128][512][512]
                 const bf16_t* __restrict__ ab1B,  // [128][128][512]
                 const bf16_t* __restrict__ ab2A,  // [128][512][128]
                 const bf16_t* __restrict__ cpB,   // [128][128][128]
                 float* __restrict__ bsum) {       // [512][128][128]
  __shared__ bf16_t As[2][64 * 32];   // 2 x 4KB
  __shared__ bf16_t Bs[2][128 * 32];  // 2 x 8KB
  int tid = threadIdx.x;
  int lane = tid & 31;
  int wv = tid >> 5;               // 0..7
  int blk = blockIdx.x;            // 128 channels * 8 i-blocks
  int c = blk >> 3;
  int i0 = (blk & 7) * 64;
  const bf16_t* A1 = ppA  + (size_t)c * 512 * 512;
  const bf16_t* B1 = ab1B + (size_t)c * 128 * 512;
  const bf16_t* A2 = ab2A + (size_t)c * 512 * 128;
  const bf16_t* B2 = cpB  + (size_t)c * 128 * 128;

  int ti = wv >> 1;                // wave's i-tile within the 64-row block
  int j0w = (wv & 1) * 64;         // wave's 4 j-tiles start here

  f32x8 acc[4];
#pragma unroll
  for (int q = 0; q < 4; ++q) acc[q] = (f32x8)0.f;

  // Cooperative staging slices: A = 16B/thread, B = 32B/thread.
  int ar = tid >> 2, ak = (tid & 3) * 8;   // As[b][ar][ak..ak+7]
  int bn = tid >> 1, bk = (tid & 1) * 16;  // Bs[b][bn][bk..bk+15]
  bf16_t* asp[2] = { &As[0][ar * 32 + ak], &As[1][ar * 32 + ak] };
  bf16_t* bsp[2] = { &Bs[0][bn * 32 + bk], &Bs[1][bn * 32 + bk] };
#if USE_ASYNC_LDS
  uint32_t asOff[2] = { (uint32_t)(uintptr_t)asp[0], (uint32_t)(uintptr_t)asp[1] };
  uint32_t bsOff[2] = { (uint32_t)(uintptr_t)bsp[0], (uint32_t)(uintptr_t)bsp[1] };
#endif

  const int NCH = 20;  // 16 chunks of block1 + 4 chunks of block2

  // Stage chunk `kc` (global source select) into LDS buffer `b`.
  auto stage = [&](int kc, int b) {
    const bf16_t* ga;
    const bf16_t* gb;
    if (kc < 16) {
      ga = A1 + (size_t)(i0 + ar) * 512 + kc * 32 + ak;
      gb = B1 + (size_t)bn * 512 + kc * 32 + bk;
    } else {
      int k2 = kc - 16;
      ga = A2 + (size_t)(i0 + ar) * 128 + k2 * 32 + ak;
      gb = B2 + (size_t)bn * 128 + k2 * 32 + bk;
    }
#if USE_ASYNC_LDS
    async_copy_b128(asOff[b], ga);
    async_copy_b128(bsOff[b], gb);
    async_copy_b128(bsOff[b] + 16, gb + 8);
#else
    *(bf16x8*)asp[b] = *(const bf16x8*)ga;
    *(bf16x8*)bsp[b] = *(const bf16x8*)gb;
    *(bf16x8*)(bsp[b] + 8) = *(const bf16x8*)(gb + 8);
#endif
  };

  // Prologue: fill buffer 0 with chunk 0.
  stage(0, 0);
#if USE_ASYNC_LDS
  wait_asynccnt0();
#endif
  __syncthreads();

#pragma unroll 1
  for (int kc = 0; kc < NCH; ++kc) {
    int cur = kc & 1;
    if (kc + 1 < NCH) stage(kc + 1, cur ^ 1);  // overlaps with compute below

    bf16x16 a = load_amat(&As[cur][0], 32, ti * 16, 0, lane);
#pragma unroll
    for (int q = 0; q < 4; ++q) {
      bf16x16 b = load_bmat(&Bs[cur][0], 32, j0w + q * 16, 0, lane);
      acc[q] = wmma_bf16(a, b, acc[q]);
    }
#if USE_ASYNC_LDS
    wait_asynccnt0();  // next chunk's copies have landed (own wave)
#endif
    __syncthreads();   // publish across waves; buffer reuse is now safe
  }

  int nlo = lane & 15, h = lane >> 4;
#pragma unroll
  for (int q = 0; q < 4; ++q) {
#pragma unroll
    for (int v = 0; v < 8; ++v) {
      int i = i0 + ti * 16 + v + 8 * h;
      int j = j0w + q * 16 + nlo;
      bsum[((size_t)i * 128 + j) * 128 + c] = acc[q][v];
    }
  }
}

// ---------------------------------------------------------------------------
// Host-side launcher
// ---------------------------------------------------------------------------
extern "C" void kernel_launch(void* const* d_in, const int* in_sizes, int n_in,
                              void* d_out, int out_size, void* d_ws, size_t ws_size,
                              hipStream_t stream) {
  (void)in_sizes; (void)n_in; (void)out_size; (void)ws_size;
  const float* z    = (const float*)d_in[0];
  const float* ppf  = (const float*)d_in[1];
  const float* cpf  = (const float*)d_in[2];
  const float* zm   = (const float*)d_in[3];   // [R] (trailing dim 1)
  const float* lnw  = (const float*)d_in[4];
  const float* lncw = (const float*)d_in[5];
  const float* gw1  = (const float*)d_in[6];
  const float* gw2  = (const float*)d_in[7];
  const float* w1   = (const float*)d_in[8];
  const float* b1   = (const float*)d_in[9];
  const float* w2   = (const float*)d_in[10];
  const float* b2   = (const float*)d_in[11];
  const float* we   = (const float*)d_in[12];
  const float* be   = (const float*)d_in[13];
  const float* wsm  = (const float*)d_in[14];
  const float* bs   = (const float*)d_in[15];
  float* out = (float*)d_out;

  const int NP = 512, NC = 128, E = 256, C = 128;
  const int Rz = NP * NC;    // 65536
  const int Rpp = NP * NP;   // 262144
  const int Rcp = NC * NC;   // 16384

  char* p = (char*)d_ws;
  auto alloc = [&](size_t bytes) -> char* {
    char* q = p;
    p += (bytes + 255) & ~(size_t)255;
    return q;
  };
  bf16_t* zb   = (bf16_t*)alloc((size_t)Rz * E * 2);
  bf16_t* ppb  = (bf16_t*)alloc((size_t)Rpp * E * 2);
  bf16_t* cpb  = (bf16_t*)alloc((size_t)Rcp * E * 2);
  bf16_t* gw1t = (bf16_t*)alloc((size_t)C * E * 2);
  bf16_t* w1t  = (bf16_t*)alloc((size_t)C * E * 2);
  bf16_t* gw2t = (bf16_t*)alloc((size_t)C * E * 2);
  bf16_t* w2t  = (bf16_t*)alloc((size_t)C * E * 2);
  bf16_t* wet  = (bf16_t*)alloc((size_t)E * E * 2);
  bf16_t* wst  = (bf16_t*)alloc((size_t)E * C * 2);
  bf16_t* ab1B = (bf16_t*)alloc((size_t)C * NC * NP * 2);
  bf16_t* ab2A = (bf16_t*)alloc((size_t)C * NP * NC * 2);
  bf16_t* ppA  = (bf16_t*)alloc((size_t)C * NP * NP * 2);
  bf16_t* cpB  = (bf16_t*)alloc((size_t)C * NC * NC * 2);
  float*  bsum = (float*)alloc((size_t)Rz * C * 4);
  bf16_t* hb   = (bf16_t*)alloc((size_t)Rz * C * 2);
  float*  gbuf = (float*)alloc((size_t)Rz * E * 4);

  // 1. LayerNorms (over E) -> bf16
  ln_rows256_bf16<<<Rz / 8, 256, 0, stream>>>(z, lnw, zb, Rz);
  ln_rows256_bf16<<<Rpp / 8, 256, 0, stream>>>(ppf, lnw, ppb, Rpp);
  ln_rows256_bf16<<<Rcp / 8, 256, 0, stream>>>(cpf, lnw, cpb, Rcp);

  // 2. Weights -> bf16 [N][K]
  cvt_w_transpose<<<(E * C + 255) / 256, 256, 0, stream>>>(gw1, gw1t, E, C);
  cvt_w_transpose<<<(E * C + 255) / 256, 256, 0, stream>>>(w1, w1t, E, C);
  cvt_w_transpose<<<(E * C + 255) / 256, 256, 0, stream>>>(gw2, gw2t, E, C);
  cvt_w_transpose<<<(E * C + 255) / 256, 256, 0, stream>>>(w2, w2t, E, C);
  cvt_w_transpose<<<(E * E + 255) / 256, 256, 0, stream>>>(we, wet, E, E);
  cvt_w_transpose<<<(C * E + 255) / 256, 256, 0, stream>>>(wsm, wst, C, E);

  // 3. Gated projections -> channel-major bf16 operands for the triangle GEMMs
  //    ab1[p,q,c] -> ab1B[c][q][p] (B operand of block1): remap r=p*128+q -> q*512+p
  gated_proj<<<Rz / 128, 256, 0, stream>>>(zb, gw1t, w1t, b1, zm, ab1B, Rz, NC, NP, 1);
  //    ab2[i,k,c] -> ab2A[c][i][k] (A operand of block2): identity remap
  gated_proj<<<Rz / 128, 256, 0, stream>>>(zb, gw2t, w2t, b2, zm, ab2A, Rz, Rz, 1, 0);
  //    pp[i,k,c]  -> ppA[c][i][k] (A operand of block1)
  gated_proj<<<Rpp / 128, 256, 0, stream>>>(ppb, gw2t, w2t, b2, nullptr, ppA, Rpp, Rpp, 1, 0);
  //    cp[j,k,c]  -> cpB[c][j][k] (B operand of block2)
  gated_proj<<<Rcp / 128, 256, 0, stream>>>(cpb, gw1t, w1t, b1, nullptr, cpB, Rcp, Rcp, 1, 0);

  // 4. g = sigmoid(z @ we + be), f32 [Rz][256]
  dense_proj<<<Rz / 128, 256, 0, stream>>>(zb, wet, be, nullptr, nullptr, gbuf,
                                           Rz, E, E, 0);

  // 5. Fused triangle einsums -> bsum [i][j][c]
  fused_block_gemm<<<128 * 8, 256, 0, stream>>>(ppA, ab1B, ab2A, cpB, bsum);

  // 6. LayerNorm over channels -> bf16 rows for the final GEMM
  ln_rows128_bf16<<<Rz / 8, 256, 0, stream>>>(bsum, lncw, hb, Rz);

  // 7. out = g * (LN_c(block1+block2) @ ws + bs) * mask
  dense_proj<<<Rz / 128, 256, 0, stream>>>(hb, wst, bs, gbuf, zm, out, Rz, E, C, 1);
}
